// GCNLayer_32074815766767
// MI455X (gfx1250) — matrix operand order repaired and checked by reading, not compiled, not run on hardware
//
#include <hip/hip_runtime.h>

#define N_NODES 16384
#define FDIM 256
#define EPS_VAL 0.0001f

typedef __attribute__((ext_vector_type(16))) __bf16 v16bf;
typedef __attribute__((ext_vector_type(8)))  float  v8f;

static __device__ __forceinline__ unsigned int f32_to_bf16_bits(float f) {
  union { float f; unsigned int u; } v; v.f = f;
  // round-to-nearest-even
  return (v.u + 0x7FFFu + ((v.u >> 16) & 1u)) >> 16;
}
static __device__ __forceinline__ unsigned int pack2_bf16(float lo, float hi) {
  return f32_to_bf16_bits(lo) | (f32_to_bf16_bits(hi) << 16);
}

// ---------------------------------------------------------------------------
// Kernel A: supportT[f][n] = bf16( sum_k features[n][k] * weight[k][f] )
// grid = (N/256, FDIM); thread n varies fastest -> coalesced 2B stores.
// weight[k][f] is uniform per block (scalar path); features row is per-thread
// sequential (cacheline-friendly).
// ---------------------------------------------------------------------------
__global__ void __launch_bounds__(256)
gcn_support_bf16T(const float* __restrict__ features,
                  const float* __restrict__ weight,
                  unsigned short* __restrict__ supportT) {
  const int n = blockIdx.x * 256 + threadIdx.x;
  const int f = blockIdx.y;
  const float* __restrict__ frow = features + (size_t)n * FDIM;
  float acc = 0.0f;
#pragma unroll 8
  for (int k = 0; k < FDIM; ++k)
    acc = fmaf(frow[k], weight[(size_t)k * FDIM + f], acc);
  supportT[(size_t)f * N_NODES + n] = (unsigned short)f32_to_bf16_bits(acc);
}

// ---------------------------------------------------------------------------
// Kernel B: out = relu( diag(1/rowsum(adj)) * (adj @ support) ) + EPS
// One block = 16 rows of adj, all 256 output columns, single pass over adj.
// 8 waves: wave w computes cols [32w, 32w+32) as two 16x16 WMMA tiles.
// ---------------------------------------------------------------------------
__global__ void __launch_bounds__(256)
gcn_propagate(const float* __restrict__ adj,
              const unsigned short* __restrict__ supportT,
              float* __restrict__ out) {
  __shared__ unsigned short aTile[16 * 128];   // bf16 A stage, row-major [16][128]
  __shared__ float rsumLds[256];
  __shared__ float invLds[16];

  const int tid   = threadIdx.x;
  const int lane  = tid & 31;
  const int wave  = tid >> 5;            // 0..7
  const int lHalf = lane >> 4;           // 0 or 1
  const int lMod  = lane & 15;

  const int rowBase = blockIdx.x * 16;
  const int colBase = wave * 32;

  // A-tile staging assignment: 256 thr load 16x128 f32 (8 floats each)
  const int ldRow = tid >> 4;            // 0..15
  const int ldK   = (tid & 15) * 8;      // 0..120
  const float* __restrict__ aSrc = adj + (size_t)(rowBase + ldRow) * N_NODES + ldK;

  union Frag { int4 i[2]; v16bf v; };

  v8f acc0 = {};
  v8f acc1 = {};
  float rsum = 0.0f;

  for (int kb = 0; kb < N_NODES; kb += 128) {
    // ---- stage A tile (f32 -> bf16) + fused rowsum accumulation ----
    const float4 f0 = *(const float4*)(aSrc + kb);
    const float4 f1 = *(const float4*)(aSrc + kb + 4);
    if (kb + 128 < N_NODES) {
      __builtin_prefetch(aSrc + kb + 128, 0, 0);   // global_prefetch next tile
    }
    rsum += (f0.x + f0.y) + (f0.z + f0.w) + (f1.x + f1.y) + (f1.z + f1.w);

    int4 packed;
    packed.x = (int)pack2_bf16(f0.x, f0.y);
    packed.y = (int)pack2_bf16(f0.z, f0.w);
    packed.z = (int)pack2_bf16(f1.x, f1.y);
    packed.w = (int)pack2_bf16(f1.z, f1.w);
    *(int4*)(aTile + ldRow * 128 + ldK) = packed;

    __syncthreads();

    // ---- 4 x K32 WMMA steps on the staged tile ----
#pragma unroll
    for (int kk = 0; kk < 4; ++kk) {
      // A fragment (16x32 bf16) per ISA layout:
      // lanes 0-15: row=lane,   K = {kk*32+0..7, kk*32+16..23}
      // lanes16-31: row=lane-16,K = {kk*32+8..15, kk*32+24..31}
      Frag a;
      const unsigned short* ap = aTile + lMod * 128 + kk * 32 + lHalf * 8;
      a.i[0] = *(const int4*)(ap);        // ds_load_b128
      a.i[1] = *(const int4*)(ap + 16);   // ds_load_b128

      // B fragments (32x16 bf16): lane = column, K = {8*lHalf+0..7, 16+8*lHalf+0..7}
      // (supportT is [FDIM][N_NODES], K-contiguous per column)
      const size_t kpos = (size_t)(kb + kk * 32 + lHalf * 8);
      Frag b0, b1;
      const int4* bp0 = (const int4*)(supportT + (size_t)(colBase + lMod) * N_NODES + kpos);
      b0.i[0] = bp0[0];
      b0.i[1] = bp0[2];
      const int4* bp1 = (const int4*)(supportT + (size_t)(colBase + 16 + lMod) * N_NODES + kpos);
      b1.i[0] = bp1[0];
      b1.i[1] = bp1[2];

      acc0 = __builtin_amdgcn_wmma_f32_16x16x32_bf16(
          false, a.v, false, b0.v, (short)0, acc0, false, false);
      acc1 = __builtin_amdgcn_wmma_f32_16x16x32_bf16(
          false, a.v, false, b1.v, (short)0, acc1, false, false);
    }
    __syncthreads();
  }

  // ---- reduce rowsums (16 partials per row) and invert ----
  rsumLds[tid] = rsum;   // thread t holds partial for row t/16
  __syncthreads();
  if (tid < 16) {
    float s = 0.0f;
#pragma unroll
    for (int j = 0; j < 16; ++j) s += rsumLds[tid * 16 + j];
    invLds[tid] = 1.0f / s;
  }
  __syncthreads();

  // ---- scale, relu, +eps, store ----
  // C/D layout: VGPR i -> row m = i + 8*lHalf ; column n = lMod (+16 for tile1)
  float inv[8];
#pragma unroll
  for (int i = 0; i < 8; ++i) inv[i] = invLds[lHalf * 8 + i];

  float* __restrict__ o0 = out + (size_t)(rowBase + lHalf * 8) * FDIM + colBase + lMod;
  float* __restrict__ o1 = o0 + 16;
#pragma unroll
  for (int i = 0; i < 8; ++i) {
    float r0 = fmaxf(acc0[i] * inv[i], 0.0f) + EPS_VAL;
    float r1 = fmaxf(acc1[i] * inv[i], 0.0f) + EPS_VAL;
    o0[(size_t)i * FDIM] = r0;
    o1[(size_t)i * FDIM] = r1;
  }
}

// ---------------------------------------------------------------------------
extern "C" void kernel_launch(void* const* d_in, const int* in_sizes, int n_in,
                              void* d_out, int out_size, void* d_ws, size_t ws_size,
                              hipStream_t stream) {
  (void)in_sizes; (void)n_in; (void)out_size; (void)ws_size;
  const float* features = (const float*)d_in[0];
  const float* adj      = (const float*)d_in[1];
  const float* weight   = (const float*)d_in[2];
  float* out            = (float*)d_out;

  // workspace: support^T as bf16, [FDIM][N_NODES] = 8 MB (fits L2 easily)
  unsigned short* supportT = (unsigned short*)d_ws;

  dim3 gridA(N_NODES / 256, FDIM);
  gcn_support_bf16T<<<gridA, 256, 0, stream>>>(features, weight, supportT);

  gcn_propagate<<<N_NODES / 16, 256, 0, stream>>>(adj, supportT, out);
}
